// MultiHeadAttention_64269890617481
// MI455X (gfx1250) — matrix operand order repaired and checked by reading, not compiled, not run on hardware
//
#include <hip/hip_runtime.h>

typedef __attribute__((ext_vector_type(16))) __bf16 v16bf;
typedef __attribute__((ext_vector_type(8)))  __bf16 bf16x8;
typedef __attribute__((ext_vector_type(2)))  __bf16 bf16x2;
typedef __attribute__((ext_vector_type(8)))  float  v8f;
typedef __attribute__((ext_vector_type(4)))  float  f32x4;

#define WMMA_BF16(a, b, c) \
  __builtin_amdgcn_wmma_f32_16x16x32_bf16(false, (a), false, (b), (short)0, (c), false, false)

static constexpr int Bsz = 4, Seq = 2048, Dm = 1024, Hn = 16, Dk = 64;
static constexpr int Mrows = Bsz * Seq;

// ---- gfx1250 async copy: global -> LDS (16B per lane), tracked by ASYNCcnt ----
__device__ inline void async_ld_b128(void* lds_ptr, const void* gptr) {
  unsigned loff = (unsigned)(uintptr_t)lds_ptr;            // addr[31:0] = LDS offset
  unsigned long long ga = (unsigned long long)(uintptr_t)gptr;
  asm volatile("global_load_async_to_lds_b128 %0, %1, off"
               :: "v"(loff), "v"(ga) : "memory");
}
__device__ inline void wait_asynccnt0() {
  asm volatile("s_wait_asynccnt 0x0" ::: "memory");
}

// ---------------- fragment loaders (wave32, 16x16x32 bf16 layouts) -------------
// A (16x32): lane r=lane&15 -> row; k = half*8 + i (i<8), 16 + half*8 + (i-8)
__device__ inline v16bf frag_a(const __bf16* base, int stride, int lane) {
  int r = lane & 15, h = lane >> 4;
  const __bf16* p = base + r * stride + h * 8;
  bf16x8 lo = *(const bf16x8*)p;
  bf16x8 hi = *(const bf16x8*)(p + 16);
  v16bf f;
#pragma unroll
  for (int i = 0; i < 8; ++i) { f[i] = lo[i]; f[i + 8] = hi[i]; }
  return f;
}
// B (32x16) from N-major (transposed) LDS: lane n=lane&15 -> col; k = half*8 + i (i<8), 16 + half*8 + (i-8)
__device__ inline v16bf frag_b(const __bf16* base, int stride, int lane) {
  int n = lane & 15, h = lane >> 4;
  const __bf16* p = base + n * stride + h * 8;
  bf16x8 lo = *(const bf16x8*)p;
  bf16x8 hi = *(const bf16x8*)(p + 16);
  v16bf f;
#pragma unroll
  for (int i = 0; i < 8; ++i) { f[i] = lo[i]; f[i + 8] = hi[i]; }
  return f;
}

// ---------------- GEMM: C[M,N] = A[M,1024] @ W[1024,N] + bias ------------------
// Register-staged double buffering; SPLIT_OUT: bf16 head-split [B,H,S,64] out.
template <bool A_BF16, bool SPLIT_OUT>
__global__ __launch_bounds__(256) void gemm_bias_kernel(
    const void* __restrict__ A_, const float* __restrict__ W,
    const float* __restrict__ bias, void* __restrict__ Cout) {
  constexpr int K = Dm, N = Dm;
  __shared__ __bf16 As[128 * 40];   // [row][k], stride 40 (bank-conflict-free)
  __shared__ __bf16 Bs[128 * 40];   // transposed: [n][k], stride 40
  int tid = threadIdx.x;
  int m0 = blockIdx.x * 128, n0 = blockIdx.y * 128;
  int wave = tid >> 5, lane = tid & 31;
  int wm = wave >> 1, wn = wave & 1;   // waves: 4 (M) x 2 (N), each 32x64

  v8f acc[2][4];
#pragma unroll
  for (int i = 0; i < 2; ++i)
#pragma unroll
    for (int j = 0; j < 4; ++j) acc[i][j] = {};

  int arow = tid >> 1, acb = (tid & 1) * 16;   // A: 128x32, 16 elems/thread
  int wk = (tid >> 4) * 2, wn8 = (tid & 15) * 8;  // W: 2 k-rows x 8 n/thread

  // staging registers
  bf16x8 a_s0, a_s1;
  f32x4 w_r[4];

  auto load_global = [&](int k0) {
    if constexpr (A_BF16) {
      const __bf16* ap = (const __bf16*)A_ + (size_t)(m0 + arow) * K + k0 + acb;
      a_s0 = *(const bf16x8*)ap;
      a_s1 = *(const bf16x8*)(ap + 8);
    } else {
      const float* ap = (const float*)A_ + (size_t)(m0 + arow) * K + k0 + acb;
      f32x4 v0 = *(const f32x4*)(ap + 0);
      f32x4 v1 = *(const f32x4*)(ap + 4);
      f32x4 v2 = *(const f32x4*)(ap + 8);
      f32x4 v3 = *(const f32x4*)(ap + 12);
#pragma unroll
      for (int e = 0; e < 4; ++e) {
        a_s0[e]     = (__bf16)v0[e];
        a_s0[e + 4] = (__bf16)v1[e];
        a_s1[e]     = (__bf16)v2[e];
        a_s1[e + 4] = (__bf16)v3[e];
      }
    }
    const float* wp = W + (size_t)(k0 + wk) * N + n0 + wn8;
    w_r[0] = *(const f32x4*)(wp + 0);
    w_r[1] = *(const f32x4*)(wp + 4);
    w_r[2] = *(const f32x4*)(wp + N + 0);
    w_r[3] = *(const f32x4*)(wp + N + 4);
  };
  auto store_lds = [&]() {
    *(bf16x8*)&As[arow * 40 + acb]     = a_s0;
    *(bf16x8*)&As[arow * 40 + acb + 8] = a_s1;
#pragma unroll
    for (int e = 0; e < 8; ++e) {   // packed (k,k+1) pair -> ds_store_b32
      float lo = (e < 4) ? w_r[0][e] : w_r[1][e - 4];
      float hi = (e < 4) ? w_r[2][e] : w_r[3][e - 4];
      bf16x2 pr; pr[0] = (__bf16)lo; pr[1] = (__bf16)hi;
      *(bf16x2*)&Bs[(wn8 + e) * 40 + wk] = pr;
    }
  };

  load_global(0);
  store_lds();
  __syncthreads();

  for (int k0 = 0; k0 < K; k0 += 32) {
    bool has_next = (k0 + 32) < K;
    if (has_next) load_global(k0 + 32);        // overlap with WMMA below
    if (k0 + 64 < K) {                         // gfx1250 global_prefetch_b8
      if constexpr (A_BF16)
        __builtin_prefetch((const __bf16*)A_ + (size_t)(m0 + arow) * K + k0 + 64 + acb, 0, 1);
      else
        __builtin_prefetch((const float*)A_ + (size_t)(m0 + arow) * K + k0 + 64 + acb, 0, 1);
      __builtin_prefetch(W + (size_t)(k0 + 64 + wk) * N + n0 + wn8, 0, 1);
    }

    v16bf a0 = frag_a(As + (32 * wm + 0) * 40, 40, lane);
    v16bf a1 = frag_a(As + (32 * wm + 16) * 40, 40, lane);
#pragma unroll
    for (int j = 0; j < 4; ++j) {
      v16bf b = frag_b(Bs + (64 * wn + 16 * j) * 40, 40, lane);
      acc[0][j] = WMMA_BF16(a0, b, acc[0][j]);
      acc[1][j] = WMMA_BF16(a1, b, acc[1][j]);
    }
    __syncthreads();                           // all reads of this tile done
    if (has_next) {
      store_lds();
      __syncthreads();                         // next tile visible
    }
  }

  // epilogue: C element v -> m = v + half*8, n = lane&15
  int half = lane >> 4, ln = lane & 15;
#pragma unroll
  for (int i = 0; i < 2; ++i) {
#pragma unroll
    for (int j = 0; j < 4; ++j) {
      int ng = n0 + 64 * wn + 16 * j + ln;
      float bv = bias[ng];
#pragma unroll
      for (int v = 0; v < 8; ++v) {
        int mg = m0 + 32 * wm + 16 * i + v + half * 8;
        float val = acc[i][j][v] + bv;
        if constexpr (SPLIT_OUT) {
          int bb = mg >> 11, s = mg & 2047;       // mg = b*Seq + s
          int hh = ng >> 6, d = ng & 63;          // ng = h*64 + d
          ((__bf16*)Cout)[(((size_t)(bb * Hn + hh) * Seq + s) << 6) + d] = (__bf16)val;
        } else {
          ((float*)Cout)[(size_t)mg * N + ng] = val;
        }
      }
    }
  }
}

// ---------------- Flash attention: per (b,h), 128 Q rows per block -------------
// K tiles: async global->LDS copies (ASYNCcnt) into a ping-pong pair.
// V tiles: register-staged (transposed on the way into LDS).
__global__ __launch_bounds__(256) void attention_kernel(
    const __bf16* __restrict__ Qbf, const __bf16* __restrict__ Kbf,
    const __bf16* __restrict__ Vbf, const float* __restrict__ Mask,
    __bf16* __restrict__ Ctx) {
  __shared__ __bf16 Ks[2][64 * 72];    // ping-pong [key][dk], stride 72
  __shared__ __bf16 Vt[64 * 72];       // transposed: [dk][key]
  __shared__ __bf16 Ps[8 * 16 * 72];   // per-wave P staging [row][key]
  int tid = threadIdx.x;
  int wave = tid >> 5, lane = tid & 31;
  int half = lane >> 4, ln = lane & 15;
  int bh = blockIdx.y;
  int b = bh >> 4, h = bh & 15;
  int q0 = blockIdx.x * 128 + wave * 16;
  const size_t headBase = (size_t)bh * Seq * Dk;

  // Q A-fragments for this wave's 16 rows (dk chunks 0..31, 32..63), from global
  v16bf qa[2];
#pragma unroll
  for (int c = 0; c < 2; ++c) {
    const __bf16* p = Qbf + headBase + (size_t)(q0 + ln) * Dk + c * 32 + half * 8;
    bf16x8 lo = *(const bf16x8*)p;
    bf16x8 hi = *(const bf16x8*)(p + 16);
    v16bf f;
#pragma unroll
    for (int i = 0; i < 8; ++i) { f[i] = lo[i]; f[i + 8] = hi[i]; }
    qa[c] = f;
  }

  float mrow[8], lrow[8];
  v8f accO[4];
#pragma unroll
  for (int v = 0; v < 8; ++v) { mrow[v] = -3.0e38f; lrow[v] = 0.f; }
#pragma unroll
  for (int t = 0; t < 4; ++t) accO[t] = {};

  int lkey = tid >> 2, ldb = (tid & 3) * 16;      // K: 1 key x 16 dk/thread
  int vkp = (tid >> 3) * 2, vdb = (tid & 7) * 8;  // V: 2 keys x 8 dk/thread
  __bf16* pw = Ps + wave * 16 * 72;

  bf16x8 vst0, vst1;   // V staging registers
  auto async_k = [&](int kt0, int buf) {
    const __bf16* kp = Kbf + headBase + (size_t)(kt0 + lkey) * Dk + ldb;
    async_ld_b128(&Ks[buf][lkey * 72 + ldb], kp);
    async_ld_b128(&Ks[buf][lkey * 72 + ldb + 8], kp + 8);
  };
  auto load_v = [&](int kt0) {
    const __bf16* vp = Vbf + headBase + (size_t)(kt0 + vkp) * Dk + vdb;
    vst0 = *(const bf16x8*)vp;
    vst1 = *(const bf16x8*)(vp + Dk);
  };
  auto store_v = [&]() {
#pragma unroll
    for (int e = 0; e < 8; ++e) {      // packed (key,key+1) -> ds_store_b32
      bf16x2 pr; pr[0] = vst0[e]; pr[1] = vst1[e];
      *(bf16x2*)&Vt[(vdb + e) * 72 + vkp] = pr;
    }
  };

  async_k(0, 0);
  load_v(0);
  store_v();
  wait_asynccnt0();
  __syncthreads();

  for (int it = 0; it < Seq / 64; ++it) {
    int kt0 = it * 64;
    int cur = it & 1;
    bool has_next = (kt0 + 64) < Seq;
    if (has_next) {                              // overlap copies with compute
      async_k(kt0 + 64, cur ^ 1);
      load_v(kt0 + 64);
    }
    if (kt0 + 128 < Seq) {                       // gfx1250 global_prefetch_b8
      __builtin_prefetch(Kbf + headBase + (size_t)(kt0 + 128 + lkey) * Dk + ldb, 0, 1);
      __builtin_prefetch(Vbf + headBase + (size_t)(kt0 + 128 + vkp) * Dk + vdb, 0, 1);
    }
    const __bf16* Kcur = Ks[cur];

    // logits S = Q @ K^T : 4 key sub-tiles x 2 dk chunks
    v8f accS[4];
#pragma unroll
    for (int j = 0; j < 4; ++j) {
      v8f s = {};
#pragma unroll
      for (int c = 0; c < 2; ++c) {
        v16bf kb = frag_b(Kcur + (j * 16) * 72 + c * 32, 72, lane);
        s = WMMA_BF16(qa[c], kb, s);
      }
      accS[j] = s;
    }

    // scale + mask + row max (rows live in 16-lane half-groups)
    float rmax[8];
#pragma unroll
    for (int v = 0; v < 8; ++v) rmax[v] = -3.0e38f;
#pragma unroll
    for (int j = 0; j < 4; ++j) {
      float mv = Mask[(size_t)b * Seq + kt0 + j * 16 + ln] * -1.0e9f;
#pragma unroll
      for (int v = 0; v < 8; ++v) {
        float x = accS[j][v] * 0.125f + mv;   // 1/sqrt(64) = 0.125
        accS[j][v] = x;
        rmax[v] = fmaxf(rmax[v], x);
      }
    }
#pragma unroll
    for (int v = 0; v < 8; ++v)
#pragma unroll
      for (int m = 1; m < 16; m <<= 1)
        rmax[v] = fmaxf(rmax[v], __shfl_xor(rmax[v], m, 32));

    // online softmax update
    float rsum[8];
#pragma unroll
    for (int v = 0; v < 8; ++v) {
      float mn = fmaxf(mrow[v], rmax[v]);
      float corr = __expf(mrow[v] - mn);
      mrow[v] = mn;
      lrow[v] *= corr;
#pragma unroll
      for (int t = 0; t < 4; ++t) accO[t][v] *= corr;
      rsum[v] = 0.f;
    }
#pragma unroll
    for (int j = 0; j < 4; ++j)
#pragma unroll
      for (int v = 0; v < 8; ++v) {
        float p = __expf(accS[j][v] - mrow[v]);
        accS[j][v] = p;
        rsum[v] += p;
      }
#pragma unroll
    for (int v = 0; v < 8; ++v) {
#pragma unroll
      for (int m = 1; m < 16; m <<= 1) rsum[v] += __shfl_xor(rsum[v], m, 32);
      lrow[v] += rsum[v];
    }

    // C-layout -> A-layout via per-wave LDS staging (LDS is in-order per wave)
#pragma unroll
    for (int j = 0; j < 4; ++j)
#pragma unroll
      for (int v = 0; v < 8; ++v)
        pw[(v + half * 8) * 72 + j * 16 + ln] = (__bf16)accS[j][v];

    v16bf pa0 = frag_a(pw + 0, 72, lane);
    v16bf pa1 = frag_a(pw + 32, 72, lane);
#pragma unroll
    for (int t = 0; t < 4; ++t) {
      v16bf vb0 = frag_b(Vt + (t * 16) * 72 + 0, 72, lane);
      accO[t] = WMMA_BF16(pa0, vb0, accO[t]);
      v16bf vb1 = frag_b(Vt + (t * 16) * 72 + 32, 72, lane);
      accO[t] = WMMA_BF16(pa1, vb1, accO[t]);
    }
    __syncthreads();                             // all reads of Vt/Ks[cur] done
    if (has_next) {
      store_v();
      wait_asynccnt0();                          // K copies into Ks[cur^1] done
      __syncthreads();                           // next tile visible to all
    }
  }

  // ctx[b][s][h*64+d] bf16 (merged heads), normalized by l
#pragma unroll
  for (int t = 0; t < 4; ++t)
#pragma unroll
    for (int v = 0; v < 8; ++v) {
      int mg = q0 + v + half * 8;
      float o = accO[t][v] / lrow[v];
      Ctx[(size_t)(b * Seq + mg) * Dm + h * 64 + t * 16 + ln] = (__bf16)o;
    }
}

// -------------------------------- launcher -------------------------------------
extern "C" void kernel_launch(void* const* d_in, const int* in_sizes, int n_in,
                              void* d_out, int out_size, void* d_ws, size_t ws_size,
                              hipStream_t stream) {
  const float* q    = (const float*)d_in[0];
  const float* k    = (const float*)d_in[1];
  const float* v    = (const float*)d_in[2];
  const float* mask = (const float*)d_in[3];
  const float* Wq = (const float*)d_in[4];  const float* bq = (const float*)d_in[5];
  const float* Wk = (const float*)d_in[6];  const float* bk = (const float*)d_in[7];
  const float* Wv = (const float*)d_in[8];  const float* bv = (const float*)d_in[9];
  const float* Wo = (const float*)d_in[10]; const float* bo = (const float*)d_in[11];

  size_t headElems = (size_t)Bsz * Hn * Seq * Dk;   // 8,388,608
  __bf16* qbf = (__bf16*)d_ws;
  __bf16* kbf = qbf + headElems;
  __bf16* vbf = kbf + headElems;
  __bf16* ctx = vbf + headElems;                    // total 64 MB of d_ws

  dim3 gg(Mrows / 128, Dm / 128);                   // 64 x 8 blocks
  gemm_bias_kernel<false, true><<<gg, 256, 0, stream>>>(q, Wq, bq, qbf);
  gemm_bias_kernel<false, true><<<gg, 256, 0, stream>>>(k, Wk, bk, kbf);
  gemm_bias_kernel<false, true><<<gg, 256, 0, stream>>>(v, Wv, bv, vbf);
  attention_kernel<<<dim3(Seq / 128, Bsz * Hn), 256, 0, stream>>>(qbf, kbf, vbf, mask, ctx);
  gemm_bias_kernel<true, false><<<gg, 256, 0, stream>>>(ctx, Wo, bo, d_out);
}
